// GraphTransformerLayer_34428457844903
// MI455X (gfx1250) — matrix operand order repaired and checked by reading, not compiled, not run on hardware
//
#include <hip/hip_runtime.h>
#include <stdint.h>
#include <stddef.h>

// ---------------------------------------------------------------------------
// Graph Transformer layer for MI455X (gfx1250, wave32, WMMA).
// Dominant cost: two dense 4096^3 matmuls (norm^2, norm^4) -> bf16 WMMA,
// f32 accumulate; operands are L2-resident (32MB bf16 << 192MB L2).
// All GEMMs are NT (C = A * W^T): both WMMA fragments are contiguous row
// reads (k = 16*(lane/16) + s, same order in A and B).
// GEMM wave tile is 32x64 (2 A-frags x 4 B-frags = 8 WMMA / k-step); the 8
// waves of a block share one B tile (same tn) so B is served from WGP$/L0
// and only the A stream hits L2 (kept ahead with global_prefetch_b8).
// attn_scatter stages K/V rows into LDS with GLOBAL_LOAD_ASYNC_TO_LDS_B64
// (ASYNCcnt) -- the CDNA5 direct memory->LDS path, no VGPR bounce.
// ---------------------------------------------------------------------------

#define NNODES 4096
#define DIMF   512
#define HIDF   1024
#define NHEAD  8
#define KNBR   20
#define NSCALE 3

typedef __bf16 bf16;
typedef bf16  bf16x16 __attribute__((ext_vector_type(16)));
typedef bf16  bf16x8  __attribute__((ext_vector_type(8)));
typedef float f32x8   __attribute__((ext_vector_type(8)));

// ------------------------------- elementwise -------------------------------

__global__ void zero_f32(float* __restrict__ p, size_t n) {
  size_t i = (size_t)blockIdx.x * blockDim.x + threadIdx.x;
  size_t st = (size_t)gridDim.x * blockDim.x;
  for (; i < n; i += st) p[i] = 0.0f;
}

__global__ void f2bf(const float* __restrict__ x, bf16* __restrict__ y, size_t n) {
  size_t i = (size_t)blockIdx.x * blockDim.x + threadIdx.x;
  size_t st = (size_t)gridDim.x * blockDim.x;
  for (; i < n; i += st) y[i] = (bf16)x[i];
}

__global__ void addk(const float* __restrict__ a, const float* __restrict__ b,
                     float* __restrict__ y, size_t n) {
  size_t i = (size_t)blockIdx.x * blockDim.x + threadIdx.x;
  size_t st = (size_t)gridDim.x * blockDim.x;
  for (; i < n; i += st) y[i] = a[i] + b[i];
}

// --------------------------- graph construction ----------------------------

__global__ void scatter_edges(const int* __restrict__ src, const int* __restrict__ dst,
                              float* __restrict__ A, int E) {
  int e = blockIdx.x * 256 + threadIdx.x;
  if (e >= E) return;
  int s = src[e], d = dst[e];
  atomicAdd(&A[(size_t)s * NNODES + d], 1.0f);
  atomicAdd(&A[(size_t)d * NNODES + s], 1.0f);   // adj + adj.T
}

__global__ void rowsum_dinv(const float* __restrict__ A, float* __restrict__ dinv) {
  int row = blockIdx.x;
  float s = 0.0f;
  for (int j = threadIdx.x; j < NNODES; j += 256) s += A[(size_t)row * NNODES + j];
  __shared__ float ls[256];
  ls[threadIdx.x] = s;
  __syncthreads();
  for (int off = 128; off; off >>= 1) {
    if ((int)threadIdx.x < off) ls[threadIdx.x] += ls[threadIdx.x + off];
    __syncthreads();
  }
  if (threadIdx.x == 0) dinv[row] = (ls[0] > 0.0f) ? rsqrtf(ls[0]) : 0.0f;
}

__global__ void normalize_bf(float* __restrict__ A, const float* __restrict__ dinv,
                             bf16* __restrict__ Ab) {
  size_t t = (size_t)blockIdx.x * 256 + threadIdx.x;   // exactly N*N threads
  int r = (int)(t >> 12);
  int c = (int)(t & 4095);
  float v = A[t] * dinv[r] * dinv[c];
  A[t] = v;
  Ab[t] = (bf16)v;
}

// ------------------------------ WMMA NT GEMM -------------------------------
// C[M,N](f32 and/or bf16) = A[M,K](bf16,row-major) * Bt[N,K](bf16,row-major)^T
// Wave tile 32(M) x 64(N), K-step 32, 8 accumulators.
// A frag (16-bit A 16x32): lane -> row m0+(lane&15); 8-elt chunks at
//   K = k0 + (lane<16?0:16) and +8. B frag (16-bit B 32x16): lane -> column
//   n0+(lane&15) == row of Bt; 16 consecutive K at k0 + (lane<16?0:16).
// C frag: VGPR r -> M = r + 8*(lane>=16), N = n0 + (lane&15).

__global__ __launch_bounds__(256) void wmma_gemm_nt(
    const bf16* __restrict__ A, const bf16* __restrict__ Bt,
    float* __restrict__ C, bf16* __restrict__ Cb,
    const float* __restrict__ bias,
    int M, int N, int Kd, int relu) {
  int wid = ((int)blockIdx.x * 256 + (int)threadIdx.x) >> 5;
  int Mt = M >> 5, Nt = N >> 6;
  if (wid >= Mt * Nt) return;               // wave-uniform: EXEC stays all-ones
  int tm = wid % Mt, tn = wid / Mt;         // tm fastest: block's 8 waves share tn
  int m0 = tm << 5, n0 = tn << 6;
  int lane = (int)threadIdx.x & 31;
  int l16 = lane & 15;
  int half = lane >> 4;

  const bf16* ap0 = A + (size_t)(m0 + l16) * Kd + half * 16;
  const bf16* ap1 = ap0 + (size_t)16 * Kd;
  const bf16* bp0 = Bt + (size_t)(n0 + l16) * Kd + half * 16;
  const bf16* bp1 = bp0 + (size_t)16 * Kd;
  const bf16* bp2 = bp0 + (size_t)32 * Kd;
  const bf16* bp3 = bp0 + (size_t)48 * Kd;

  f32x8 acc[2][4];
#pragma unroll
  for (int a = 0; a < 2; ++a)
#pragma unroll
    for (int b = 0; b < 4; ++b) acc[a][b] = (f32x8){0, 0, 0, 0, 0, 0, 0, 0};

  for (int k0 = 0; k0 < Kd; k0 += 32) {
    union { bf16x16 v; bf16x8 h[2]; } af0, af1;
    af0.h[0] = *(const bf16x8*)(ap0 + k0);
    af0.h[1] = *(const bf16x8*)(ap0 + k0 + 8);
    af1.h[0] = *(const bf16x8*)(ap1 + k0);
    af1.h[1] = *(const bf16x8*)(ap1 + k0 + 8);
    bf16x16 b0 = *(const bf16x16*)(bp0 + k0);
    bf16x16 b1 = *(const bf16x16*)(bp1 + k0);
    bf16x16 b2 = *(const bf16x16*)(bp2 + k0);
    bf16x16 b3 = *(const bf16x16*)(bp3 + k0);
    // keep the A stream (the only real L2 stream) ahead of the consumer
    __builtin_prefetch(ap0 + k0 + 512, 0, 1);
#ifdef __AMDGCN__
    acc[0][0] = __builtin_amdgcn_wmma_f32_16x16x32_bf16(false, af0.v, false, b0, (short)0, acc[0][0], false, false);
    acc[1][0] = __builtin_amdgcn_wmma_f32_16x16x32_bf16(false, af1.v, false, b0, (short)0, acc[1][0], false, false);
    acc[0][1] = __builtin_amdgcn_wmma_f32_16x16x32_bf16(false, af0.v, false, b1, (short)0, acc[0][1], false, false);
    acc[1][1] = __builtin_amdgcn_wmma_f32_16x16x32_bf16(false, af1.v, false, b1, (short)0, acc[1][1], false, false);
    acc[0][2] = __builtin_amdgcn_wmma_f32_16x16x32_bf16(false, af0.v, false, b2, (short)0, acc[0][2], false, false);
    acc[1][2] = __builtin_amdgcn_wmma_f32_16x16x32_bf16(false, af1.v, false, b2, (short)0, acc[1][2], false, false);
    acc[0][3] = __builtin_amdgcn_wmma_f32_16x16x32_bf16(false, af0.v, false, b3, (short)0, acc[0][3], false, false);
    acc[1][3] = __builtin_amdgcn_wmma_f32_16x16x32_bf16(false, af1.v, false, b3, (short)0, acc[1][3], false, false);
#endif
  }

#pragma unroll
  for (int a = 0; a < 2; ++a) {
    int mb = m0 + a * 16 + half * 8;
#pragma unroll
    for (int b = 0; b < 4; ++b) {
      int n = n0 + b * 16 + l16;
      float bb = bias ? bias[n] : 0.0f;
#pragma unroll
      for (int r = 0; r < 8; ++r) {
        float v = acc[a][b][r] + bb;
        if (relu) v = fmaxf(v, 0.0f);
        if (C) C[(size_t)(mb + r) * N + n] = v;
        if (Cb) Cb[(size_t)(mb + r) * N + n] = (bf16)v;
      }
    }
  }
}

// --------------------------------- top-k -----------------------------------
// One wave per row: per-lane register top-20 (statically indexed, predicated
// insertion), then 20 rounds of cross-lane shfl_xor argmax. Tie -> lower idx.

__global__ __launch_bounds__(256) void topk20(const float* __restrict__ A,
                                              int* __restrict__ out) {
  int row = blockIdx.x * 8 + ((int)threadIdx.x >> 5);
  int lane = (int)threadIdx.x & 31;
  const float* a = A + (size_t)row * NNODES;

  float v[KNBR];
  int ix[KNBR];
#pragma unroll
  for (int p = 0; p < KNBR; ++p) { v[p] = -__builtin_inff(); ix[p] = 0x7fffffff; }

  for (int j = lane; j < NNODES; j += 32) {
    float cv = (j == row) ? 0.0f : a[j];    // fill_diagonal_(0)
    int ci = j;
    bool act = (cv > v[0]) || (cv == v[0] && ci < ix[0]);
    if (act) {
      float cur_v = cv; int cur_i = ci;
#pragma unroll
      for (int p = 0; p < KNBR; ++p) {
        bool shift = act && (p < KNBR - 1) &&
                     ((cur_v > v[p + 1]) || (cur_v == v[p + 1] && cur_i < ix[p + 1]));
        if (act) { v[p] = shift ? v[p + 1] : cur_v; ix[p] = shift ? ix[p + 1] : cur_i; }
        act = shift;
      }
    }
  }

  for (int r = 0; r < KNBR; ++r) {
    float bv = -__builtin_inff(); int bi = 0x7fffffff; int bs = -1;
#pragma unroll
    for (int p = 0; p < KNBR; ++p) {
      bool bet = (v[p] > bv) || (v[p] == bv && ix[p] < bi);
      if (bet) { bv = v[p]; bi = ix[p]; bs = p; }
    }
    float rv = bv; int ri = bi; int rl = lane;
#pragma unroll
    for (int off = 16; off; off >>= 1) {
      float ov = __shfl_xor(rv, off);
      int oi = __shfl_xor(ri, off);
      int ol = __shfl_xor(rl, off);
      bool bet = (ov > rv) || (ov == rv && oi < ri);
      if (bet) { rv = ov; ri = oi; rl = ol; }
    }
    if (lane == rl) {
#pragma unroll
      for (int p = 0; p < KNBR; ++p) if (p == bs) v[p] = -__builtin_inff();
    }
    if (lane == 0) out[row * KNBR + r] = ri;
  }
}

// ------------------------------- attention ---------------------------------
// K/V rows staged into LDS via CDNA5 async memory->LDS loads (ASYNCcnt):
// each of the 256 threads issues one GLOBAL_LOAD_ASYNC_TO_LDS_B64 per array
// (2 floats), then waits its own ASYNCcnt and hits the workgroup barrier,
// after which all waves' LDS writes are visible.

__global__ __launch_bounds__(256) void attn_scatter(
    const float* __restrict__ Qh, const float* __restrict__ Kh,
    const float* __restrict__ Vh, const int* __restrict__ idx,
    float* __restrict__ wV, float* __restrict__ z) {
  __shared__ float sK[DIMF];
  __shared__ float sV[DIMF];
  int i = blockIdx.x;
  {
    int t2 = (int)threadIdx.x * 2;                 // 256 threads x 2 floats = 512
    const float* gk = Kh + (size_t)i * DIMF + t2;
    const float* gv = Vh + (size_t)i * DIMF + t2;
#ifdef __AMDGCN__
    unsigned lk = (unsigned)(size_t)&sK[t2];       // low 32 bits = LDS byte offset
    unsigned lv = (unsigned)(size_t)&sV[t2];
    asm volatile("global_load_async_to_lds_b64 %0, %1, off"
                 :: "v"(lk), "v"(gk) : "memory");
    asm volatile("global_load_async_to_lds_b64 %0, %1, off"
                 :: "v"(lv), "v"(gv) : "memory");
    asm volatile("s_wait_asynccnt 0x0" ::: "memory");
#else
    sK[t2] = gk[0]; sK[t2 + 1] = gk[1];
    sV[t2] = gv[0]; sV[t2 + 1] = gv[1];
#endif
  }
  __syncthreads();
  int head = (int)threadIdx.x >> 5;   // 8 waves == 8 heads
  int lane = (int)threadIdx.x & 31;
  for (int j = 0; j < KNBR; ++j) {
    int dn = idx[i * KNBR + j];
    const float* q = Qh + (size_t)dn * DIMF + head * 64;
    float part = sK[head * 64 + lane] * q[lane] + sK[head * 64 + 32 + lane] * q[32 + lane];
#pragma unroll
    for (int off = 16; off; off >>= 1) part += __shfl_xor(part, off);
    float sc = __expf(fminf(fmaxf(part * 0.125f, -5.0f), 5.0f));  // /sqrt(64), clip
    float* wv = wV + (size_t)dn * DIMF + head * 64;
    atomicAdd(&wv[lane],      sc * sV[head * 64 + lane]);
    atomicAdd(&wv[32 + lane], sc * sV[head * 64 + 32 + lane]);
    if (lane == 0) atomicAdd(&z[(size_t)dn * NHEAD + head], sc);
  }
}

__global__ void scale_softmax(const float* __restrict__ h,
                              const float* __restrict__ W1, const float* __restrict__ b1,
                              const float* __restrict__ W2, const float* __restrict__ b2,
                              float* __restrict__ sw) {
  int i = blockIdx.x * 8 + ((int)threadIdx.x >> 5);
  int lane = (int)threadIdx.x & 31;
  float acc[16];
#pragma unroll
  for (int k = 0; k < 16; ++k) acc[k] = 0.0f;
  const float* hr = h + (size_t)i * DIMF;
  for (int d = lane; d < DIMF; d += 32) {
    float hv = hr[d];
#pragma unroll
    for (int k = 0; k < 16; ++k) acc[k] += hv * W1[k * DIMF + d];
  }
#pragma unroll
  for (int k = 0; k < 16; ++k) {
#pragma unroll
    for (int off = 16; off; off >>= 1) acc[k] += __shfl_xor(acc[k], off);
  }
  if (lane == 0) {
    float lg[NSCALE];
#pragma unroll
    for (int s = 0; s < NSCALE; ++s) lg[s] = b2[s];
#pragma unroll
    for (int k = 0; k < 16; ++k) {
      float hv = fmaxf(acc[k] + b1[k], 0.0f);
#pragma unroll
      for (int s = 0; s < NSCALE; ++s) lg[s] += hv * W2[s * 16 + k];
    }
    float mx = fmaxf(lg[0], fmaxf(lg[1], lg[2]));
    float e0 = __expf(lg[0] - mx), e1 = __expf(lg[1] - mx), e2 = __expf(lg[2] - mx);
    float inv = 1.0f / (e0 + e1 + e2);
    sw[i * 3 + 0] = e0 * inv; sw[i * 3 + 1] = e1 * inv; sw[i * 3 + 2] = e2 * inv;
  }
}

__global__ void combine_accum(const float* __restrict__ wV, const float* __restrict__ z,
                              const float* __restrict__ sw, int s,
                              float* __restrict__ attn) {
  size_t t = (size_t)blockIdx.x * 256 + threadIdx.x;   // N*DIM threads exactly
  int i = (int)(t >> 9);
  int f = (int)(t & 511);
  int hh = f >> 6;
  attn[t] += sw[i * 3 + s] * wV[t] / (z[(size_t)i * NHEAD + hh] + 1e-6f);
}

// ------------------------------- batchnorm ---------------------------------

__global__ void bn_stats(const float* __restrict__ x, const float* __restrict__ g,
                         const float* __restrict__ b, float* __restrict__ scale,
                         float* __restrict__ shift) {
  int c = blockIdx.x;  // DIM columns
  float s = 0.0f, sq = 0.0f;
  for (int r = threadIdx.x; r < NNODES; r += 256) {
    float v = x[(size_t)r * DIMF + c];
    s += v; sq += v * v;
  }
  __shared__ float ls[256], lq[256];
  ls[threadIdx.x] = s; lq[threadIdx.x] = sq;
  __syncthreads();
  for (int off = 128; off; off >>= 1) {
    if ((int)threadIdx.x < off) { ls[threadIdx.x] += ls[threadIdx.x + off]; lq[threadIdx.x] += lq[threadIdx.x + off]; }
    __syncthreads();
  }
  if (threadIdx.x == 0) {
    float m = ls[0] * (1.0f / NNODES);
    float var = lq[0] * (1.0f / NNODES) - m * m;       // biased var
    float sc = g[c] * rsqrtf(var + 1e-5f);
    scale[c] = sc;
    shift[c] = b[c] - m * sc;
  }
}

__global__ void bn_apply(const float* __restrict__ x, const float* __restrict__ scale,
                         const float* __restrict__ shift, float* __restrict__ y,
                         bf16* __restrict__ yb) {
  size_t t = (size_t)blockIdx.x * 256 + threadIdx.x;   // N*DIM threads exactly
  int c = (int)(t & 511);
  float v = x[t] * scale[c] + shift[c];
  y[t] = v;
  if (yb) yb[t] = (bf16)v;
}

// --------------------------------- driver ----------------------------------

static void launch_gemm(const bf16* A, const bf16* Bt, float* C, bf16* Cb,
                        const float* bias, int M, int N, int K, int relu,
                        hipStream_t s) {
  int waves = (M >> 5) * (N >> 6);
  int blocks = (waves + 7) / 8;
  wmma_gemm_nt<<<blocks, 256, 0, s>>>(A, Bt, C, Cb, bias, M, N, K, relu);
}

extern "C" void kernel_launch(void* const* d_in, const int* in_sizes, int n_in,
                              void* d_out, int out_size, void* d_ws, size_t ws_size,
                              hipStream_t stream) {
  const float* h    = (const float*)d_in[0];
  const float* Wq   = (const float*)d_in[1];
  const float* Wk   = (const float*)d_in[2];
  const float* Wv   = (const float*)d_in[3];
  const float* Wo   = (const float*)d_in[4];
  const float* Wsw1 = (const float*)d_in[5];
  const float* bsw1 = (const float*)d_in[6];
  const float* Wsw2 = (const float*)d_in[7];
  const float* bsw2 = (const float*)d_in[8];
  const float* Wf1  = (const float*)d_in[9];
  const float* bf1  = (const float*)d_in[10];
  const float* Wf2  = (const float*)d_in[11];
  const float* bf2  = (const float*)d_in[12];
  const float* bn1g = (const float*)d_in[13];
  const float* bn1b = (const float*)d_in[14];
  const float* bn2g = (const float*)d_in[15];
  const float* bn2b = (const float*)d_in[16];
  const int*   src  = (const int*)d_in[17];
  const int*   dst  = (const int*)d_in[18];
  int E = in_sizes[17];
  float* out = (float*)d_out;
  (void)n_in; (void)out_size; (void)ws_size;

  const size_t NN2 = (size_t)NNODES * NNODES;
  const size_t ND  = (size_t)NNODES * DIMF;
  const size_t NHd = (size_t)NNODES * HIDF;

  uintptr_t w = (uintptr_t)d_ws;
  auto take = [&](size_t bytes) -> void* {
    uintptr_t p = w;
    w = (w + bytes + 255) & ~(uintptr_t)255;
    return (void*)p;
  };

  // survives both phases
  int* idx = (int*)take((size_t)NSCALE * NNODES * KNBR * sizeof(int));
  uintptr_t mark = w;

  // ---------------- phase 1: diffusion graph (matrix powers + top-k) -------
  float* Pa   = (float*)take(NN2 * 4);   // adj -> norm f32 -> P4
  float* Pb   = (float*)take(NN2 * 4);   // P2
  bf16*  Bh   = (bf16*) take(NN2 * 2);   // norm bf16 -> P2 bf16
  float* dinv = (float*)take(NNODES * 4);

  zero_f32<<<4096, 256, 0, stream>>>(Pa, NN2);
  scatter_edges<<<(E + 255) / 256, 256, 0, stream>>>(src, dst, Pa, E);
  rowsum_dinv<<<NNODES, 256, 0, stream>>>(Pa, dinv);
  normalize_bf<<<(unsigned)(NN2 / 256), 256, 0, stream>>>(Pa, dinv, Bh);

  topk20<<<NNODES / 8, 256, 0, stream>>>(Pa, idx + 0 * NNODES * KNBR);           // t=1
  launch_gemm(Bh, Bh, Pb, nullptr, nullptr, NNODES, NNODES, NNODES, 0, stream);  // P2
  f2bf<<<8192, 256, 0, stream>>>(Pb, Bh, NN2);
  topk20<<<NNODES / 8, 256, 0, stream>>>(Pb, idx + 1 * NNODES * KNBR);           // t=2
  launch_gemm(Bh, Bh, Pa, nullptr, nullptr, NNODES, NNODES, NNODES, 0, stream);  // P4
  topk20<<<NNODES / 8, 256, 0, stream>>>(Pa, idx + 2 * NNODES * KNBR);           // t=4

  // ---------------- phase 2: transformer layer (reuse phase-1 space) -------
  w = mark;
  bf16*  hbf   = (bf16*) take(ND * 2);
  bf16*  wbuf  = (bf16*) take((size_t)HIDF * DIMF * 2);
  float* Qh    = (float*)take(ND * 4);
  float* Khb   = (float*)take(ND * 4);
  float* Vhb   = (float*)take(ND * 4);
  float* wVb   = (float*)take(ND * 4);
  float* zb    = (float*)take((size_t)NNODES * NHEAD * 4);
  float* swb   = (float*)take((size_t)NNODES * NSCALE * 4);
  float* attnf = (float*)take(ND * 4);
  bf16*  attnb = (bf16*) take(ND * 2);
  float* tmp   = (float*)take(ND * 4);
  float* res   = (float*)take(ND * 4);
  float* h1    = (float*)take(ND * 4);
  bf16*  h1b   = (bf16*) take(ND * 2);
  bf16*  hidb  = (bf16*) take(NHd * 2);
  float* sc1   = (float*)take(DIMF * 4);
  float* sh1   = (float*)take(DIMF * 4);
  float* sc2   = (float*)take(DIMF * 4);
  float* sh2   = (float*)take(DIMF * 4);

  f2bf<<<4096, 256, 0, stream>>>(h, hbf, ND);
  scale_softmax<<<NNODES / 8, 256, 0, stream>>>(h, Wsw1, bsw1, Wsw2, bsw2, swb);
  zero_f32<<<4096, 256, 0, stream>>>(attnf, ND);

  for (int s = 0; s < NSCALE; ++s) {
    f2bf<<<1024, 256, 0, stream>>>(Wq + (size_t)s * DIMF * DIMF, wbuf, (size_t)DIMF * DIMF);
    launch_gemm(hbf, wbuf, Qh, nullptr, nullptr, NNODES, DIMF, DIMF, 0, stream);
    f2bf<<<1024, 256, 0, stream>>>(Wk + (size_t)s * DIMF * DIMF, wbuf, (size_t)DIMF * DIMF);
    launch_gemm(hbf, wbuf, Khb, nullptr, nullptr, NNODES, DIMF, DIMF, 0, stream);
    f2bf<<<1024, 256, 0, stream>>>(Wv + (size_t)s * DIMF * DIMF, wbuf, (size_t)DIMF * DIMF);
    launch_gemm(hbf, wbuf, Vhb, nullptr, nullptr, NNODES, DIMF, DIMF, 0, stream);

    zero_f32<<<4096, 256, 0, stream>>>(wVb, ND);
    zero_f32<<<128, 256, 0, stream>>>(zb, (size_t)NNODES * NHEAD);
    attn_scatter<<<NNODES, 256, 0, stream>>>(Qh, Khb, Vhb, idx + s * NNODES * KNBR, wVb, zb);
    combine_accum<<<8192, 256, 0, stream>>>(wVb, zb, swb, s, attnf);
  }

  // output projection + residual + BN1
  f2bf<<<4096, 256, 0, stream>>>(attnf, attnb, ND);
  f2bf<<<1024, 256, 0, stream>>>(Wo, wbuf, (size_t)DIMF * DIMF);
  launch_gemm(attnb, wbuf, tmp, nullptr, nullptr, NNODES, DIMF, DIMF, 0, stream);
  addk<<<4096, 256, 0, stream>>>(h, tmp, res, ND);
  bn_stats<<<DIMF, 256, 0, stream>>>(res, bn1g, bn1b, sc1, sh1);
  bn_apply<<<8192, 256, 0, stream>>>(res, sc1, sh1, h1, h1b);

  // FFN + residual + BN2 (hidden GEMM writes bf16 directly from epilogue)
  f2bf<<<2048, 256, 0, stream>>>(Wf1, wbuf, (size_t)HIDF * DIMF);
  launch_gemm(h1b, wbuf, nullptr, hidb, bf1, NNODES, HIDF, DIMF, 1, stream);  // relu fused
  f2bf<<<2048, 256, 0, stream>>>(Wf2, wbuf, (size_t)DIMF * HIDF);
  launch_gemm(hidb, wbuf, tmp, nullptr, bf2, NNODES, DIMF, HIDF, 0, stream);
  addk<<<4096, 256, 0, stream>>>(h1, tmp, res, ND);
  bn_stats<<<DIMF, 256, 0, stream>>>(res, bn2g, bn2b, sc2, sh2);
  bn_apply<<<8192, 256, 0, stream>>>(res, sc2, sh2, out, (bf16*)nullptr);
}
